// head_colbert_like_42047729828546
// MI455X (gfx1250) — compile-verified, matrix operand order checked by reading, not run on hardware
//
#include <hip/hip_runtime.h>

#define DEV __device__ __forceinline__

typedef __attribute__((ext_vector_type(16))) __bf16 v16bf;
typedef __attribute__((ext_vector_type(8)))  __bf16 v8bf;
typedef __attribute__((ext_vector_type(8)))  float  v8f;

// Problem constants (match reference)
constexpr int Bn = 16, SQ = 1024, SC = 1024, Dd = 768, Hh = 1536;

DEV unsigned short f2bf(float f) {
  unsigned u = __builtin_bit_cast(unsigned, f);
  u += 0x7FFFu + ((u >> 16) & 1u);          // round-to-nearest-even
  return (unsigned short)(u >> 16);
}

DEV v8bf ldb8(const __bf16* p) { return *reinterpret_cast<const v8bf*>(p); }
DEV v16bf cat16(v8bf lo, v8bf hi) {
  return __builtin_shufflevector(lo, hi, 0,1,2,3,4,5,6,7,8,9,10,11,12,13,14,15);
}

// ---------------------------------------------------------------- K0: fp32 -> bf16
__global__ __launch_bounds__(256) void k_convert(
    const float* __restrict__ q, const float* __restrict__ c,
    unsigned short* __restrict__ qb, unsigned short* __restrict__ cb, int n) {
  int i = blockIdx.x * blockDim.x + threadIdx.x;
  int stride = gridDim.x * blockDim.x;
  for (; i < n; i += stride) { qb[i] = f2bf(q[i]); cb[i] = f2bf(c[i]); }
}

// ---------------------------------------------------------------- K1: S = Q @ C^T (bf16 WMMA) + mask
// One wave computes a 32(M) x 64(N) tile: 2 A frags x 4 B frags -> 8 WMMAs / k-step.
// ~21 FLOP per byte loaded (all operands L2-resident).
__global__ __launch_bounds__(256) void k_qkt_wmma(
    const unsigned short* __restrict__ qb16, const unsigned short* __restrict__ cb16,
    const int* __restrict__ mask_q, const int* __restrict__ mask_c,
    float* __restrict__ s) {
  const int lane  = threadIdx.x & 31;
  const int wave  = blockIdx.x * (blockDim.x >> 5) + (threadIdx.x >> 5);
  const int b     = wave >> 9;             // 32 * 16 tiles per batch
  const int rem   = wave & 511;
  const int mt    = rem >> 4;              // 0..31  (M tile of 32)
  const int ng    = rem & 15;              // 0..15  (N group of 64)
  const int row16 = lane & 15;
  const int kg    = lane >> 4;             // K-half select per ISA A/B layout

  const __bf16* qp0 = reinterpret_cast<const __bf16*>(qb16) +
                      ((size_t)b * SQ + mt * 32 + row16) * Dd;
  const __bf16* qp1 = qp0 + (size_t)16 * Dd;
  const __bf16* cpb = reinterpret_cast<const __bf16*>(cb16) +
                      ((size_t)b * SC + ng * 64 + row16) * Dd;

  v8f acc00 = {}, acc01 = {}, acc02 = {}, acc03 = {};
  v8f acc10 = {}, acc11 = {}, acc12 = {}, acc13 = {};
  for (int k0 = 0; k0 < Dd; k0 += 32) {
    const int ka = k0 + kg * 8;
    v16bf a0 = cat16(ldb8(qp0 + ka),                ldb8(qp0 + ka + 16));
    v16bf a1 = cat16(ldb8(qp1 + ka),                ldb8(qp1 + ka + 16));
    v16bf f0 = cat16(ldb8(cpb + 0 * 16 * Dd + ka),  ldb8(cpb + 0 * 16 * Dd + ka + 16));
    v16bf f1 = cat16(ldb8(cpb + 1 * 16 * Dd + ka),  ldb8(cpb + 1 * 16 * Dd + ka + 16));
    v16bf f2 = cat16(ldb8(cpb + 2 * 16 * Dd + ka),  ldb8(cpb + 2 * 16 * Dd + ka + 16));
    v16bf f3 = cat16(ldb8(cpb + 3 * 16 * Dd + ka),  ldb8(cpb + 3 * 16 * Dd + ka + 16));
    acc00 = __builtin_amdgcn_wmma_f32_16x16x32_bf16(false, a0, false, f0, (short)0, acc00, false, false);
    acc01 = __builtin_amdgcn_wmma_f32_16x16x32_bf16(false, a0, false, f1, (short)0, acc01, false, false);
    acc02 = __builtin_amdgcn_wmma_f32_16x16x32_bf16(false, a0, false, f2, (short)0, acc02, false, false);
    acc03 = __builtin_amdgcn_wmma_f32_16x16x32_bf16(false, a0, false, f3, (short)0, acc03, false, false);
    acc10 = __builtin_amdgcn_wmma_f32_16x16x32_bf16(false, a1, false, f0, (short)0, acc10, false, false);
    acc11 = __builtin_amdgcn_wmma_f32_16x16x32_bf16(false, a1, false, f1, (short)0, acc11, false, false);
    acc12 = __builtin_amdgcn_wmma_f32_16x16x32_bf16(false, a1, false, f2, (short)0, acc12, false, false);
    acc13 = __builtin_amdgcn_wmma_f32_16x16x32_bf16(false, a1, false, f3, (short)0, acc13, false, false);
  }

  // C/D layout: lanes 0-15 -> N=lane, M=r; lanes 16-31 -> N=lane-16, M=r+8
  const int colL   = lane & 15;
  const int rowoff = (lane >> 4) * 8;
  const int* mqb = mask_q + b * SQ + mt * 32;
  const int* mcb = mask_c + b * SC + ng * 64;
  float* sb = s + ((size_t)b * SQ + mt * 32) * SC + ng * 64;
  v8f accs[2][4] = {{acc00, acc01, acc02, acc03}, {acc10, acc11, acc12, acc13}};
#pragma unroll
  for (int R = 0; R < 2; R++) {
#pragma unroll
    for (int t = 0; t < 4; t++) {
      const int cc  = t * 16 + colL;
      const int mcv = mcb[cc];
#pragma unroll
      for (int r = 0; r < 8; r++) {
        const int m = R * 16 + r + rowoff;
        const bool live = (mqb[m] != 0) && (mcv != 0);
        sb[(size_t)m * SC + cc] = live ? accs[R][t][r] : -__builtin_inff();
      }
    }
  }
}

// ---------------------------------------------------------------- K2: row softmax stats (wave/row)
__global__ __launch_bounds__(256) void k_rowstats(
    const float* __restrict__ s, float* __restrict__ rowm, float* __restrict__ rowinvz) {
  const int lane = threadIdx.x & 31;
  const int row  = blockIdx.x * (blockDim.x >> 5) + (threadIdx.x >> 5); // b*SQ + i
  const float* sr = s + (size_t)row * SC;
  float m = -__builtin_inff();
  for (int j = lane; j < SC; j += 32) m = fmaxf(m, sr[j]);
  for (int off = 16; off; off >>= 1) m = fmaxf(m, __shfl_xor(m, off, 32));
  float z = 0.f;
  if (m > -__builtin_inff()) {
    for (int j = lane; j < SC; j += 32) z += __expf(sr[j] - m);
    for (int off = 16; off; off >>= 1) z += __shfl_xor(z, off, 32);
  }
  if (lane == 0) { rowm[row] = m; rowinvz[row] = (z > 0.f) ? 1.f / z : 0.f; }
}

// ---------------------------------------------------------------- K3: column softmax stats (thread/col)
__global__ __launch_bounds__(256) void k_colstats(
    const float* __restrict__ s, float* __restrict__ colm, float* __restrict__ colinvz) {
  const int cg = blockIdx.x * blockDim.x + threadIdx.x;  // b*SC + j
  const int b  = cg >> 10, j = cg & 1023;
  const float* sb = s + ((size_t)b << 20) + j;
  float m = -__builtin_inff();
  for (int i = 0; i < SQ; i++) m = fmaxf(m, sb[(size_t)i * SC]);
  float z = 0.f;
  if (m > -__builtin_inff())
    for (int i = 0; i < SQ; i++) z += __expf(sb[(size_t)i * SC] - m);
  colm[cg] = m; colinvz[cg] = (z > 0.f) ? 1.f / z : 0.f;
}

// ---------------------------------------------------------------- K4: colsum_p[b,j] = sum_i p[b,i,j]
__global__ __launch_bounds__(256) void k_colsum(
    const float* __restrict__ s, const float* __restrict__ rowm,
    const float* __restrict__ rowinvz, float* __restrict__ colsum) {
  const int cg = blockIdx.x * blockDim.x + threadIdx.x;
  const int b  = cg >> 10, j = cg & 1023;
  const float* sb = s + ((size_t)b << 20) + j;
  const float* rm = rowm + b * SQ;
  const float* rz = rowinvz + b * SQ;
  float acc = 0.f;
  for (int i = 0; i < SQ; i++) {
    const float iz = rz[i];
    if (iz > 0.f) acc += __expf(sb[(size_t)i * SC] - rm[i]) * iz;
  }
  colsum[cg] = acc;
}

// ---------------------------------------------------------------- K5: rowsum_pp[b,i] = sum_j pp[b,i,j]
__global__ __launch_bounds__(256) void k_rowsum(
    const float* __restrict__ s, const float* __restrict__ colm,
    const float* __restrict__ colinvz, float* __restrict__ rowsum) {
  const int lane = threadIdx.x & 31;
  const int row  = blockIdx.x * (blockDim.x >> 5) + (threadIdx.x >> 5); // b*SQ + i
  const int b    = row >> 10;
  const float* sr = s + (size_t)row * SC;
  const float* cm = colm + b * SC;
  const float* cz = colinvz + b * SC;
  float acc = 0.f;
  for (int j = lane; j < SC; j += 32) {
    const float iz = cz[j];
    if (iz > 0.f) acc += __expf(sr[j] - cm[j]) * iz;
  }
  for (int off = 16; off; off >>= 1) acc += __shfl_xor(acc, off, 32);
  if (lane == 0) rowsum[row] = acc;
}

// ---------------------------------------------------------------- K6: a/bb combine -> x[b, 0:1536]
__global__ __launch_bounds__(256) void k_combine(
    const float* __restrict__ q, const float* __restrict__ c,
    const float* __restrict__ colsum, const float* __restrict__ rowsum,
    float* __restrict__ x) {
  const int t = blockIdx.x * blockDim.x + threadIdx.x;  // Bn*Dd threads
  const int b = t / Dd, k = t % Dd;
  const float* cp = c + ((size_t)b * SC) * Dd + k;
  const float* qp = q + ((size_t)b * SQ) * Dd + k;
  const float* cs = colsum + b * SC;
  const float* rs = rowsum + b * SQ;
  float a = 0.f, bb = 0.f;
  for (int j = 0; j < SC; j++) {
    a  += cs[j] * cp[(size_t)j * Dd];
    bb += rs[j] * qp[(size_t)j * Dd];
  }
  x[b * Hh + k]      = a;
  x[b * Hh + Dd + k] = bb;
}

// ---------------------------------------------------------------- K7: fc1 + tanh
__global__ __launch_bounds__(256) void k_fc1(
    const float* __restrict__ x, const float* __restrict__ W1,
    const float* __restrict__ b1, float* __restrict__ xt) {
  const int t = blockIdx.x * blockDim.x + threadIdx.x;  // Bn*Hh threads
  const int b = t & 15, h = t >> 4;
  const float* xr = x + b * Hh;
  const float* w  = W1 + (size_t)h * Hh;
  float acc = b1[h];
  for (int k = 0; k < Hh; k++) acc += xr[k] * w[k];
  xt[b * Hh + h] = tanhf(acc);
}

// ---------------------------------------------------------------- K8: fc2 + log_softmax + NLL loss
__global__ void k_head(const float* __restrict__ xt, const float* __restrict__ W2,
                       const float* __restrict__ b2, const int* __restrict__ labels,
                       float* __restrict__ out) {
  __shared__ float lg[Bn][2];
  const int t = threadIdx.x;
  if (t < Bn * 2) {
    const int b = t >> 1, n = t & 1;
    const float* xr = xt + b * Hh;
    const float* w  = W2 + n * Hh;
    float acc = b2[n];
    for (int k = 0; k < Hh; k++) acc += xr[k] * w[k];
    lg[b][n] = acc;
  }
  __syncthreads();
  if (t < Bn) {
    const float l0 = lg[t][0], l1 = lg[t][1];
    const float m  = fmaxf(l0, l1);
    const float lse = m + __logf(__expf(l0 - m) + __expf(l1 - m));
    lg[t][0] = l0 - lse; lg[t][1] = l1 - lse;
    out[1 + t * 2 + 0] = l0 - lse;
    out[1 + t * 2 + 1] = l1 - lse;
  }
  __syncthreads();
  if (t == 0) {
    float loss = 0.f;
    for (int b = 0; b < Bn; b++) loss -= lg[b][labels[b]];
    out[0] = loss / (float)Bn;
  }
}

// ================================================================ launch
extern "C" void kernel_launch(void* const* d_in, const int* in_sizes, int n_in,
                              void* d_out, int out_size, void* d_ws, size_t ws_size,
                              hipStream_t stream) {
  const float* q      = (const float*)d_in[0];
  const float* c      = (const float*)d_in[1];
  const int*   mask_q = (const int*)d_in[2];
  const int*   mask_c = (const int*)d_in[3];
  const int*   labels = (const int*)d_in[4];
  const float* W1     = (const float*)d_in[5];
  const float* b1     = (const float*)d_in[6];
  const float* W2     = (const float*)d_in[7];
  const float* b2     = (const float*)d_in[8];
  float* out = (float*)d_out;

  char* ws = (char*)d_ws;
  size_t off = 0;
  auto carve = [&](size_t bytes) -> void* {
    off = (off + 255) & ~(size_t)255;
    void* p = ws + off;
    off += bytes;
    return p;
  };
  const size_t nQ = (size_t)Bn * SQ * Dd;            // 12.58M elems
  unsigned short* qb = (unsigned short*)carve(nQ * 2);
  unsigned short* cb = (unsigned short*)carve(nQ * 2);
  float* s       = (float*)carve((size_t)Bn * SQ * SC * 4);   // 67 MB
  float* rowm    = (float*)carve((size_t)Bn * SQ * 4);
  float* rowinvz = (float*)carve((size_t)Bn * SQ * 4);
  float* colm    = (float*)carve((size_t)Bn * SC * 4);
  float* colinvz = (float*)carve((size_t)Bn * SC * 4);
  float* colsum  = (float*)carve((size_t)Bn * SC * 4);
  float* rowsum  = (float*)carve((size_t)Bn * SQ * 4);
  float* x       = (float*)carve((size_t)Bn * Hh * 4);
  float* xt      = (float*)carve((size_t)Bn * Hh * 4);

  // K0: convert q,c to bf16
  k_convert<<<4096, 256, 0, stream>>>(q, c, qb, cb, (int)nQ);

  // K1: WMMA GEMM + mask; 8192 waves, 32x64 tile per wave (8 waves / block)
  k_qkt_wmma<<<1024, 256, 0, stream>>>(qb, cb, mask_q, mask_c, s);

  // K2/K3: softmax stats
  k_rowstats<<<2048, 256, 0, stream>>>(s, rowm, rowinvz);       // 16384 waves
  k_colstats<<<64,   256, 0, stream>>>(s, colm, colinvz);       // 16384 threads

  // K4/K5: probability mass reductions
  k_colsum<<<64,   256, 0, stream>>>(s, rowm, rowinvz, colsum);
  k_rowsum<<<2048, 256, 0, stream>>>(s, colm, colinvz, rowsum);

  // K6: a / bb combine -> x
  k_combine<<<(Bn * Dd) / 256, 256, 0, stream>>>(q, c, colsum, rowsum, x);

  // K7: fc1 + tanh
  k_fc1<<<(Bn * Hh) / 256, 256, 0, stream>>>(x, W1, b1, xt);

  // K8: fc2 + log_softmax + loss
  k_head<<<1, 64, 0, stream>>>(xt, W2, b2, labels, out);
}
